// HomoMSELoss_20547123544859
// MI455X (gfx1250) — matrix operand order repaired and checked by reading, not compiled, not run on hardware
//
#include <hip/hip_runtime.h>
#include <hip/hip_bf16.h>

// Pipeline: warp -> erode(+norm) -> nms x2 -> radix top-k union masks ->
// mask-mul (f16 outputs + mask output) -> WMMA gaussian + fused loss -> finalize.
// Gaussian 5x5 separable blur as banded-matrix GEMMs on v_wmma_f32_16x16x32_f16.
// Halo staging uses global_load_async_to_lds_b32 when available (ASYNCcnt path).

#define HH   512
#define WW   512
#define HWn  (HH * WW)        // 262144
#define NB   32
#define NHW  (NB * HWn)       // 8388608
#define TOPK 512

#define XSTR 40               // halo tile row stride in halves (80B, 16B aligned)
#define TSTR 40               // transposed-T column stride in halves

typedef __attribute__((ext_vector_type(16))) _Float16 v16h;
typedef __attribute__((ext_vector_type(8)))  _Float16 v8h;
typedef __attribute__((ext_vector_type(4)))  _Float16 v4h;
typedef __attribute__((ext_vector_type(8)))  float    v8f;

typedef __attribute__((address_space(1))) int gas_int;  // global AS pointer elem
typedef __attribute__((address_space(3))) int las_int;  // LDS AS pointer elem

#if __has_builtin(__builtin_amdgcn_global_load_async_to_lds_b32)
#define HAVE_ASYNC_LDS 1
#else
#define HAVE_ASYNC_LDS 0
#endif

// gauss(5, sigma=1) normalized
__device__ __constant__ float c_g[5] = {
    0.05448868f, 0.24420134f, 0.40261996f, 0.24420134f, 0.05448868f};

// ---------------------------------------------------------------- init
__global__ void k_init(unsigned* __restrict__ u1, unsigned* __restrict__ u2,
                       float* __restrict__ sAcc, float* __restrict__ normAcc) {
  int i = blockIdx.x * blockDim.x + threadIdx.x;   // grid covers HWn
  u1[i] = 0u;
  u2[i] = 0u;
  if (i == 0) { sAcc[0] = 0.f; normAcc[0] = 0.f; }
}

// ---------------------------------------------------------------- warp
__global__ void k_warp(const float* __restrict__ det2,
                       const float* __restrict__ homo,
                       float* __restrict__ wdet2, float* __restrict__ v0) {
  int idx = blockIdx.x * blockDim.x + threadIdx.x;   // grid == NHW
  int n  = idx >> 18;
  int p  = idx & (HWn - 1);
  int py = p >> 9, px = p & 511;
  const float* h = homo + n * 9;
  float X = (float)px, Y = (float)py;
  float zz = h[6] * X + h[7] * Y + h[8];
  float inv = 1.0f / zz;
  float x = (h[0] * X + h[1] * Y + h[2]) * inv;
  float y = (h[3] * X + h[4] * Y + h[5]) * inv;
  float fx = floorf(x), fy = floorf(y);
  float wx = x - fx, wy = y - fy;
  int x0 = (int)fx, y0 = (int)fy;
  const float* img = det2 + (size_t)n * HWn;
  // gfx1250 prefetch path (global_prefetch_b8)
  __builtin_prefetch(img + min(p + WW, HWn - 1), 0, 0);
  float wgt[4] = {(1.f - wx) * (1.f - wy), wx * (1.f - wy),
                  (1.f - wx) * wy,         wx * wy};
  const int dx[4] = {0, 1, 0, 1}, dy[4] = {0, 0, 1, 1};
  float acc = 0.f, ones = 0.f;
#pragma unroll
  for (int c = 0; c < 4; ++c) {
    int cx = x0 + dx[c], cy = y0 + dy[c];
    bool valid = (cx >= 0) & (cx < WW) & (cy >= 0) & (cy < HH);
    int ccx = min(max(cx, 0), WW - 1), ccy = min(max(cy, 0), HH - 1);
    float v = img[ccy * WW + ccx];
    if (valid) { acc += v * wgt[c]; ones += wgt[c]; }
  }
  wdet2[idx] = acc;
  v0[idx] = (ones > 0.f) ? 1.f : 0.f;
}

// ---------------------------------------------------------------- erode + norm
__global__ void k_erode(const float* __restrict__ v0, float* __restrict__ vis,
                        float* __restrict__ normAcc) {
  int idx = blockIdx.x * blockDim.x + threadIdx.x;
  int n  = idx >> 18;
  int p  = idx & (HWn - 1);
  int py = p >> 9, px = p & 511;
  const float* src = v0 + (size_t)n * HWn;
  float cnt = 0.f;
#pragma unroll
  for (int dy = -2; dy <= 2; ++dy)
#pragma unroll
    for (int dx = -2; dx <= 2; ++dx) {
      int yy = py + dy, xx = px + dx;
      if ((unsigned)yy < HH && (unsigned)xx < WW) cnt += src[yy * WW + xx];
    }
  float v = (cnt >= 24.5f) ? 1.f : 0.f;   // k*k - 0.5
  vis[idx] = v;
  float s = v;
#pragma unroll
  for (int o = 16; o > 0; o >>= 1) s += __shfl_down(s, o, 32);
  __shared__ float red[8];
  int lane = threadIdx.x & 31, wv = threadIdx.x >> 5;
  if (lane == 0) red[wv] = s;
  __syncthreads();
  if (threadIdx.x == 0) {
    float t = 0.f;
#pragma unroll
    for (int i = 0; i < 8; ++i) t += red[i];
    atomicAdd(normAcc, t);
  }
}

// ---------------------------------------------------------------- NMS 5x5
__global__ void k_nms(const float* __restrict__ in, float* __restrict__ out) {
  int idx = blockIdx.x * blockDim.x + threadIdx.x;
  int n  = idx >> 18;
  int p  = idx & (HWn - 1);
  int py = p >> 9, px = p & 511;
  const float* src = in + (size_t)n * HWn;
  float v = src[p];
  float m = -3.4e38f;
#pragma unroll
  for (int dy = -2; dy <= 2; ++dy)
#pragma unroll
    for (int dx = -2; dx <= 2; ++dx) {
      int yy = py + dy, xx = px + dx;
      if ((unsigned)yy < HH && (unsigned)xx < WW) m = fmaxf(m, src[yy * WW + xx]);
    }
  out[idx] = ((v == m) & (v > 0.f)) ? v : 0.f;
}

// ---------------------------------------------------------------- top-k union mask
__global__ void k_topk(const float* __restrict__ nms1,
                       const float* __restrict__ nms2,
                       unsigned* __restrict__ u1, unsigned* __restrict__ u2) {
  const float* in = (blockIdx.x < NB) ? nms1 : nms2;
  unsigned* u     = (blockIdx.x < NB) ? u1 : u2;
  int n = blockIdx.x & (NB - 1);
  const float* row = in + (size_t)n * HWn;

  __shared__ unsigned hist[2048];
  __shared__ unsigned thrBin, remain, admit;
  for (int i = threadIdx.x; i < 2048; i += blockDim.x) hist[i] = 0u;
  __syncthreads();
  for (int i = threadIdx.x; i < HWn; i += blockDim.x) {
    unsigned b = __float_as_uint(row[i]) >> 21;
    atomicAdd(&hist[b], 1u);
  }
  __syncthreads();
  if (threadIdx.x == 0) {
    unsigned cum = 0; int bsel = 0;
    for (int b = 2047; b >= 0; --b) {
      unsigned c = hist[b];
      if (cum + c >= TOPK) { bsel = b; break; }
      cum += c;
    }
    thrBin = (unsigned)bsel;
    remain = TOPK - cum;
    admit  = 0u;
  }
  __syncthreads();
  unsigned tb = thrBin, rem = remain;
  for (int i = threadIdx.x; i < HWn; i += blockDim.x) {
    unsigned b = __float_as_uint(row[i]) >> 21;
    if (b > tb) u[i] = 1u;
    else if (b == tb) {
      if (atomicAdd(&admit, 1u) < rem) u[i] = 1u;
    }
  }
}

// ---------------------------------------------------------------- mask multiply
// Writes the masked score maps as f16 (blur stage reads them; halves traffic).
__global__ void k_maskmul(const float* __restrict__ n1, const float* __restrict__ n2,
                          const unsigned* __restrict__ u1,
                          const unsigned* __restrict__ u2,
                          _Float16* __restrict__ m1h, _Float16* __restrict__ m2h,
                          float* __restrict__ maskOut) {
  int idx = blockIdx.x * blockDim.x + threadIdx.x;
  int p = idx & (HWn - 1);
  float f1 = u1[p] ? 1.f : 0.f;
  float f2 = u2[p] ? 1.f : 0.f;
  m1h[idx] = (_Float16)(n1[idx] * f1);
  m2h[idx] = (_Float16)(n2[idx] * f2);
  maskOut[idx] = f2;
}

// ---------------------------------------------------------------- WMMA gaussian + loss
// One wave per 16x16 output tile, both maps per wave.
// Row blur: T(20x16) = X(20x20) x Band(20x16)  [two M=16 WMMAs]
// Col blur: D(16x16) = BandT(16x20) x T(20x16)
// All LDS fragment accesses are contiguous 16B-aligned runs (ds_load_b128).
__global__ void __launch_bounds__(256) k_gauss_loss(
    const _Float16* __restrict__ m1h, const _Float16* __restrict__ m2h,
    const float* __restrict__ vis, float* __restrict__ sAcc) {
  __shared__ __align__(16) _Float16 Xs[8][2][20 * XSTR]; // per-wave halo tiles
  __shared__ __align__(16) _Float16 Ts[8][16 * TSTR];    // transposed intermediate

  int lane = threadIdx.x & 31;
  int wv   = threadIdx.x >> 5;
  int tile = blockIdx.x * 8 + wv;         // 32768 tiles total
  int n  = tile >> 10;
  int t  = tile & 1023;
  int ty = t >> 5, tx = t & 31;
  int r0 = ty * 16 - 2, c0 = tx * 16 - 2;  // c0 is even

  const _Float16* src[2] = {m1h + (size_t)n * HWn, m2h + (size_t)n * HWn};

  // pre-zero this wave's staging + T regions (padding / border zeros)
  {
    unsigned* xz = (unsigned*)&Xs[wv][0][0];     // 2*20*XSTR/2 = 800 dwords
    for (int i = lane; i < 800; i += 32) xz[i] = 0u;
    unsigned* tz = (unsigned*)&Ts[wv][0];        // 16*TSTR/2 = 320 dwords
    for (int i = lane; i < 320; i += 32) tz[i] = 0u;
  }
  __syncthreads();

  // stage 20x20 halo (f16). Pairs (2 halves) are either fully in or fully out.
#if HAVE_ASYNC_LDS
  for (int mi = 0; mi < 2; ++mi) {
    for (int i = lane; i < 200; i += 32) {       // 20 rows x 10 pairs
      int rr = i / 10, j = i % 10;
      int gr = r0 + rr, gc = c0 + 2 * j;
      if ((unsigned)gr < HH && (unsigned)gc < WW) {
        unsigned long long ga =
            (unsigned long long)(const void*)(src[mi] + gr * WW + gc);
        unsigned la = (unsigned)(unsigned long long)(const void*)
                          (&Xs[wv][mi][rr * XSTR + 2 * j]);
        __builtin_amdgcn_global_load_async_to_lds_b32(
            (gas_int*)ga,
            (las_int*)(unsigned long long)la,
            0, 0);
      }
    }
  }
#if __has_builtin(__builtin_amdgcn_s_wait_asynccnt)
  __builtin_amdgcn_s_wait_asynccnt(0);
#else
  asm volatile("s_wait_asynccnt 0x0" ::: "memory");
#endif
#else
  for (int mi = 0; mi < 2; ++mi) {
    for (int i = lane; i < 400; i += 32) {
      int rr = i / 20, cc = i % 20;
      int gr = r0 + rr, gc = c0 + cc;
      _Float16 v = (_Float16)0.f;
      if ((unsigned)gr < HH && (unsigned)gc < WW) v = src[mi][gr * WW + gc];
      Xs[wv][mi][rr * XSTR + cc] = v;
    }
  }
#endif
  __syncthreads();

  int mrow = lane & 15;
  int hi   = (lane >> 4) & 1;

  // B-operand band matrix for row blur: B(k,n) = g[k-n], k in [n, n+4]
  v16h bBand = {};
#pragma unroll
  for (int v = 0; v < 8; ++v)
#pragma unroll
    for (int j = 0; j < 2; ++j) {
      int k = (v < 4 ? 2 * v : 16 + 2 * (v - 4)) + (hi ? 8 : 0) + j;
      int d = k - mrow;
      float gv = (d >= 0 && d <= 4 && k < 20) ? c_g[d] : 0.f;
      bBand[2 * v + j] = (_Float16)gv;
    }
  // A-operand band matrix for column blur: A(m,k) = g[k-m], k in [m, m+4]
  v16h aBand = {};
#pragma unroll
  for (int i = 0; i < 8; ++i) {
    int kb = 2 * i + (hi ? 16 : 0);
#pragma unroll
    for (int j = 0; j < 2; ++j) {
      int k = kb + j;
      int d = k - mrow;
      float gv = (d >= 0 && d <= 4 && k < 20) ? c_g[d] : 0.f;
      aBand[2 * i + j] = (_Float16)gv;
    }
  }

  v8f dmap[2];
#pragma unroll
  for (int mi = 0; mi < 2; ++mi) {
    const _Float16* X = &Xs[wv][mi][0];
    // A fragments: two contiguous 16B runs per fragment (ds_load_b128)
    auto makeA = [&](int row) -> v16h {
      v8h lo = *(const v8h*)(X + row * XSTR + hi * 8);
      v8h hh = *(const v8h*)(X + row * XSTR + 16 + hi * 8);
      v16h a;
#pragma unroll
      for (int i = 0; i < 8; ++i) { a[i] = lo[i]; a[8 + i] = hh[i]; }
      return a;
    };
    v16h a1 = makeA(mrow);
    v16h a2 = makeA(mrow + 4);

    v8f z = {};
    v8f tTop = __builtin_amdgcn_wmma_f32_16x16x32_f16(false, a1, false, bBand,
                                                      (short)0, z, false, false);
    v8f tBot = __builtin_amdgcn_wmma_f32_16x16x32_f16(false, a2, false, bBand,
                                                      (short)0, z, false, false);

    // store T transposed: Ts[col][k]; contiguous runs -> b128/b64 stores
    _Float16* T = &Ts[wv][0];
    {
      v8h st;
#pragma unroll
      for (int i = 0; i < 8; ++i) st[i] = (_Float16)tTop[i];   // rows hi*8 + i
      *(v8h*)(T + mrow * TSTR + hi * 8) = st;
    }
    if (hi) {
      v4h st2;
#pragma unroll
      for (int i = 0; i < 4; ++i) st2[i] = (_Float16)tBot[4 + i]; // rows 16..19
      *(v4h*)(T + mrow * TSTR + 16) = st2;
    }
    __syncthreads();

    // B fragment from transposed T: contiguous 16 halves at column mrow
    v16h bT;
    {
      v8h lo = *(const v8h*)(T + mrow * TSTR + hi * 16);
      v8h hh = *(const v8h*)(T + mrow * TSTR + hi * 16 + 8);
#pragma unroll
      for (int i = 0; i < 8; ++i) { bT[i] = lo[i]; bT[8 + i] = hh[i]; }
    }
    dmap[mi] = __builtin_amdgcn_wmma_f32_16x16x32_f16(false, aBand, false, bT,
                                                      (short)0, z, false, false);
    __syncthreads();   // T reused by next map
  }

  // fused loss accumulation: sum((d1-d2)^2 * vis)
  const float* visn = vis + (size_t)n * HWn;
  int colg = tx * 16 + mrow;
  float acc = 0.f;
#pragma unroll
  for (int i = 0; i < 8; ++i) {
    int rowg = ty * 16 + (hi ? 8 : 0) + i;
    float diff = dmap[0][i] - dmap[1][i];
    acc += diff * diff * visn[rowg * WW + colg];
  }
#pragma unroll
  for (int o = 16; o > 0; o >>= 1) acc += __shfl_down(acc, o, 32);
  if (lane == 0) atomicAdd(sAcc, acc);
}

// ---------------------------------------------------------------- finalize
__global__ void k_final(const float* __restrict__ sAcc,
                        const float* __restrict__ normAcc,
                        float* __restrict__ loss) {
  if (threadIdx.x == 0 && blockIdx.x == 0) loss[0] = sAcc[0] / normAcc[0];
}

// ---------------------------------------------------------------- launch
extern "C" void kernel_launch(void* const* d_in, const int* in_sizes, int n_in,
                              void* d_out, int out_size, void* d_ws, size_t ws_size,
                              hipStream_t stream) {
  const float* det1 = (const float*)d_in[0];
  const float* det2 = (const float*)d_in[1];
  const float* homo = (const float*)d_in[2];

  float* out     = (float*)d_out;
  float* lossOut = out;            // [0]
  float* maskOut = out + 1;        // [1 .. NHW]
  float* visOut  = out + 1 + NHW;  // [1+NHW .. 1+2*NHW]

  float*    A    = (float*)d_ws;          // warped det2
  float*    B    = A + NHW;               // v0, later nms2
  float*    C    = B + NHW;               // nms1
  unsigned* u1   = (unsigned*)(C + NHW);  // union mask map1
  unsigned* u2   = u1 + HWn;              // union mask map2
  float*    sAcc = (float*)(u2 + HWn);
  float*    nAcc = sAcc + 1;
  _Float16* m1h  = (_Float16*)(sAcc + 8); // masked map1, f16
  _Float16* m2h  = m1h + NHW;             // masked map2, f16

  dim3 blk(256);
  const int gNHW = NHW / 256;  // 32768
  const int gHW  = HWn / 256;  // 1024

  k_init<<<gHW, blk, 0, stream>>>(u1, u2, sAcc, nAcc);
  k_warp<<<gNHW, blk, 0, stream>>>(det2, homo, A, B);
  k_erode<<<gNHW, blk, 0, stream>>>(B, visOut, nAcc);
  k_nms<<<gNHW, blk, 0, stream>>>(det1, C);
  k_nms<<<gNHW, blk, 0, stream>>>(A, B);
  k_topk<<<2 * NB, blk, 0, stream>>>(C, B, u1, u2);
  k_maskmul<<<gNHW, blk, 0, stream>>>(C, B, u1, u2, m1h, m2h, maskOut);
  k_gauss_loss<<<(NB * 1024) / 8, blk, 0, stream>>>(m1h, m2h, visOut, sAcc);
  k_final<<<1, 64, 0, stream>>>(sAcc, nAcc, lossOut);
}
